// mLSTM_41274635715161
// MI455X (gfx1250) — matrix operand order repaired and checked by reading, not compiled, not run on hardware
//
#include <hip/hip_runtime.h>
#include <math.h>

// ---------------------------------------------------------------------------
// mLSTM (B=8,T=1024,D=H=128,L=2) for gfx1250 (MI455X), wave32 + WMMA bf16.
// Chunkwise-parallel scan (S=16) + swizzled LDS fragment staging +
// async global->LDS bulk copies (ASYNCcnt).
// ---------------------------------------------------------------------------

typedef __bf16 bh16;
typedef __attribute__((ext_vector_type(16))) __bf16 v16bf;
typedef __attribute__((ext_vector_type(8)))  float  v8f;

static __device__ __forceinline__ v8f wmma_bf16(v16bf a, v16bf b, v8f c) {
  return __builtin_amdgcn_wmma_f32_16x16x32_bf16(
      false, a, false, b, (short)0, c, false, false);
}

// Contiguous 32B fragment load: lowers to 2x ds_load_b128.
static __device__ __forceinline__ v16bf frag_ld(const bh16* p) {
  return *(const v16bf*)p;
}

// Fragment slot maps (inverse of the ISA VGPR layout tables, wave32):
// A (16x32 MxK, bf16): lane = m + 16*((k>>3)&1), half = (k&7) + 8*(k>>4)
// B (32x16 KxN, bf16): lane = n + 16*((k>>3)&1), half = (k&7) + 8*(k>>4)
// C/D (16x16 f32):     vgpr g, lane l -> row = g + (l<16?0:8), col = l&15

static __device__ __forceinline__ unsigned lds_addr(const void* p) {
  return (unsigned)(unsigned long long)p;  // LDS offset = low 32 bits of flat
}
// Async bulk copy: 16 bytes per active lane, global(SADDR+voff) -> LDS.
static __device__ __forceinline__ void async_copy16(unsigned lds, const void* gbase,
                                                    int byteoff) {
  asm volatile("global_load_async_to_lds_b128 %0, %1, %2"
               :: "v"(lds), "v"(byteoff), "s"(gbase) : "memory");
}
static __device__ __forceinline__ void async_wait0() {
  asm volatile("s_wait_asynccnt 0" ::: "memory");
}

// ---------------------------------------------------------------------------
// Projection GEMM: Y(M x 128) = act(X(M x 128) @ W(128 x 128) + bias)
// act: 0 = identity, 1 = exp. 16-row M-tile per block; wave w -> N-tile w.
// ---------------------------------------------------------------------------
__global__ __launch_bounds__(256) void proj_wmma(
    const float* __restrict__ X, const float* __restrict__ W,
    const float* __restrict__ bias, float* __restrict__ Y, int act) {
  __shared__ alignas(32) bh16 sXA[4][32][16];     // A-staged X tile
  __shared__ alignas(32) bh16 sWB[4][8][32][16];  // B-staged W, per (kk, ntile)
  __shared__ float sB[128];
  const int tid = threadIdx.x;
  const size_t row0 = (size_t)blockIdx.x * 16;

  for (int idx = tid; idx < 16 * 128; idx += 256) {
    const int m = idx >> 7, k = idx & 127, kl = k & 31;
    sXA[k >> 5][m + 16 * ((kl >> 3) & 1)][(kl & 7) + 8 * (kl >> 4)] =
        (bh16)X[(row0 + (size_t)m) * 128 + k];
  }
  for (int idx = tid; idx < 128 * 128; idx += 256) {
    const int k = idx >> 7, n = idx & 127, kl = k & 31;
    sWB[k >> 5][n >> 4][(n & 15) + 16 * ((kl >> 3) & 1)][(kl & 7) + 8 * (kl >> 4)] = (bh16)W[idx];
  }
  if (tid < 128) sB[tid] = bias[tid];
  __syncthreads();

  const int w = tid >> 5, lane = tid & 31;
  v8f acc = {};
#pragma unroll
  for (int kk = 0; kk < 4; ++kk)
    acc = wmma_bf16(frag_ld(&sXA[kk][lane][0]), frag_ld(&sWB[kk][w][lane][0]), acc);

  const int col = 16 * w + (lane & 15);
  const int rb  = (lane < 16) ? 0 : 8;
  const float bv = sB[col];
#pragma unroll
  for (int g = 0; g < 8; ++g) {
    float vv = acc[g] + bv;
    if (act) vv = expf(vv);
    Y[(row0 + (size_t)(rb + g)) * 128 + col] = vv;
  }
}

// ---------------------------------------------------------------------------
// Chunkwise mLSTM matrix-memory scan. Grid: (batch, 16-col slab of C).
// Wave w owns C rows [16w,16w+16) x block's 16 columns as one f32 WMMA
// accumulator fragment, persistent across all 64 chunks.
// ---------------------------------------------------------------------------
__global__ __launch_bounds__(256) void mlstm_chunk_scan(
    const float* __restrict__ Q, const float* __restrict__ K,
    const float* __restrict__ V, const float* __restrict__ Ig,
    const float* __restrict__ Fg, float* __restrict__ attn, int T) {
  const int H = 128;
  const int b  = blockIdx.x;
  const int jb = blockIdx.y;                // column slab [16*jb, 16*jb+16)
  const size_t base = (size_t)b * T * H;
  const int tid = threadIdx.x, w = tid >> 5, lane = tid & 31;
  const int rb = (lane < 16) ? 0 : 8;

  // f32 streaming buffers, filled by async copies each chunk
  __shared__ alignas(16) float sQf[16][128];
  __shared__ alignas(16) float sVf[16][128];
  __shared__ alignas(16) float sIf[16][128];
  __shared__ alignas(16) float sFf[16][128];
  __shared__ alignas(16) float sKf[16][16];
  // fragment-staged bf16 operands
  __shared__ alignas(32) bh16 sQhA[4][32][16];  // qhat as A, K=i (4 slabs)
  __shared__ alignas(32) bh16 sVhB[4][32][16];  // vhat as B, K=i, N=s
  __shared__ alignas(32) bh16 sVhA[8][32][16];  // vhat^T as A per wave, K=s (pad 32)
  __shared__ alignas(32) bh16 sKB[32][16];      // K slab as B, K=s (pad 32)
  __shared__ alignas(32) bh16 sCB[4][32][16];   // C0 snapshot as B, K=i
  __shared__ alignas(32) bh16 sGA[32][16];      // masked G as A, K=s (pad 32)
  __shared__ float sCum16[128];

  // zero-fill K-dim padding slots once (k>=16 positions; never rewritten)
  for (int idx = tid; idx < 32 * 8; idx += 256)       // sGA halves 8..15
    sGA[idx >> 3][8 + (idx & 7)] = (bh16)0.f;
  for (int idx = tid; idx < 8 * 32 * 8; idx += 256)   // sVhA halves 8..15
    sVhA[idx >> 8][(idx >> 3) & 31][8 + (idx & 7)] = (bh16)0.f;
  for (int idx = tid; idx < 32 * 8; idx += 256)       // sKB halves 8..15
    sKB[idx >> 3][8 + (idx & 7)] = (bh16)0.f;

  const float* Qp = Q + base;
  const float* Kp = K + base;
  const float* Vp = V + base;
  const float* Ip = Ig + base;
  const float* Fp = Fg + base;

  v8f state = {};                    // this wave's 16x16 slab of C (f32)
  v8f acc = {};                      // wave0: attn accumulator (lives across barrier)
  const int nchunks = T / 16;

  for (int c = 0; c < nchunks; ++c) {
    const int c0 = c * 16;
    const int cb = c0 * H * 4;       // chunk byte offset within the sequence

    // ---- async bulk copy of this chunk's streams into LDS ----------------
    {
      const unsigned lq = lds_addr(&sQf[0][0]);
      const unsigned lv = lds_addr(&sVf[0][0]);
      const unsigned li = lds_addr(&sIf[0][0]);
      const unsigned lf = lds_addr(&sFf[0][0]);
      for (int u = tid; u < 512; u += 256) {   // 16x128 f32 = 512 float4
        const int bo = u * 16;
        async_copy16(lq + bo, Qp, cb + bo);
        async_copy16(lv + bo, Vp, cb + bo);
        async_copy16(li + bo, Ip, cb + bo);
        async_copy16(lf + bo, Fp, cb + bo);
      }
      if (tid < 64) {                          // 16x16 f32 K slab
        const int r = tid >> 2, seg = tid & 3;
        async_copy16(lds_addr(&sKf[r][seg * 4]), Kp,
                     cb + r * H * 4 + jb * 64 + seg * 16);
      }
      async_wait0();
    }
    __syncthreads();

    // ---- stage 1: snapshot C0; cum/qhat/vhat chains; K slab staging ------
#pragma unroll
    for (int g = 0; g < 8; ++g) {              // C0 -> B-staged bf16 snapshot
      const int i = 16 * w + rb + g, kl = i & 31;
      sCB[i >> 5][(lane & 15) + 16 * ((kl >> 3) & 1)][(kl & 7) + 8 * (kl >> 4)] = (bh16)state[g];
    }
    if (tid < 128) {
      const int i = tid, kk = i >> 5, kl = i & 31;
      const int aLane = 16 * ((kl >> 3) & 1);  // + m
      const int aHalf = (kl & 7) + 8 * (kl >> 4);
      const int bLane = 16 * ((kl >> 3) & 1);  // + n
      const int bHalf = (kl & 7) + 8 * (kl >> 4);
      float cum = 1.f;
#pragma unroll
      for (int r = 0; r < 16; ++r) {
        cum *= sFf[r][i];
        const float qh = sQf[r][i] * cum;
        const float vh = sIf[r][i] * sVf[r][i] / cum;
        sQhA[kk][r + aLane][aHalf] = (bh16)qh;              // A: (m=r, k=i)
        sVhB[kk][r + bLane][bHalf] = (bh16)vh;              // B: (k=i, n=r)
        sVhA[i >> 4][(i & 15) + 16 * ((r >> 3) & 1)][r & 7] = (bh16)vh; // A: (m=i%16, k=r)
      }
      sCum16[i] = cum;
    } else {
      const int t2 = tid - 128;
      for (int idx = t2; idx < 256; idx += 128) {
        const int s = idx >> 4, n = idx & 15;
        sKB[n + 16 * ((s >> 3) & 1)][s & 7] = (bh16)sKf[s][n];  // B: (k=s, n), s<16
      }
    }
    __syncthreads();

    // ---- stage 2a: wave0 -> Qhat@C0 (inter); wave1 -> G (intra) ----------
    if (w == 0) {
      acc = (v8f){};
#pragma unroll
      for (int kk = 0; kk < 4; ++kk)
        acc = wmma_bf16(frag_ld(&sQhA[kk][lane][0]), frag_ld(&sCB[kk][lane][0]), acc);
    } else if (w == 1) {
      v8f gacc = {};
#pragma unroll
      for (int kk = 0; kk < 4; ++kk)
        gacc = wmma_bf16(frag_ld(&sQhA[kk][lane][0]), frag_ld(&sVhB[kk][lane][0]), gacc);
#pragma unroll
      for (int g = 0; g < 8; ++g) {           // causal mask, A-staged (k=s<16)
        const int t = rb + g, s = lane & 15;
        const float gv = (s <= t) ? gacc[g] : 0.f;
        sGA[t + 16 * ((s >> 3) & 1)][s & 7] = (bh16)gv;
      }
    }
    __syncthreads();

    // ---- stage 2b: attn store (wave0) + state update (all waves) ---------
    if (w == 0) {
      acc = wmma_bf16(frag_ld(&sGA[lane][0]), frag_ld(&sKB[lane][0]), acc);
#pragma unroll
      for (int g = 0; g < 8; ++g)
        attn[base + (size_t)(c0 + rb + g) * H + 16 * jb + (lane & 15)] = acc[g];
    }
    // C_next = cum16 (row) * (C0 + VhatT @ Kc); WMMA accumulates into C0.
    state = wmma_bf16(frag_ld(&sVhA[w][lane][0]), frag_ld(&sKB[lane][0]), state);
#pragma unroll
    for (int g = 0; g < 8; ++g)
      state[g] *= sCum16[16 * w + rb + g];
    __syncthreads();   // protect staged operands before next chunk's writes
  }
}

// ---------------------------------------------------------------------------
// Output-gate recurrence: h_t = sigmoid(h_{t-1} @ Wo + bo) * attn_t.
// One block per batch; Wo columns live in registers (64 per lane-half).
// ---------------------------------------------------------------------------
__global__ __launch_bounds__(256) void gate_scan(
    const float* __restrict__ attn, const float* __restrict__ Wo,
    const float* __restrict__ bo, float* __restrict__ hout, int T) {
  const int H = 128;
  const int b = blockIdx.x;
  const int tid = threadIdx.x;
  const int j = tid & 127;
  const int half = tid >> 7;
  const size_t base = (size_t)b * T * H;
  __shared__ float sh[128];
  __shared__ float spart[128];
  float wcol[64];
#pragma unroll
  for (int r = 0; r < 64; ++r) wcol[r] = Wo[(size_t)(64 * half + r) * H + j];
  const float bj = bo[j];
  if (tid < 128) sh[tid] = 0.f;
  __syncthreads();
  for (int t = 0; t < T; ++t) {
    float s = 0.f;
#pragma unroll
    for (int r = 0; r < 64; ++r) s += wcol[r] * sh[64 * half + r];
    if (half == 1) spart[j] = s;
    __syncthreads();
    if (half == 0) {
      const float tot = s + spart[j] + bj;
      const float o   = 1.f / (1.f + expf(-tot));
      const float hn  = o * attn[base + (size_t)t * H + j];
      sh[j] = hn;
      hout[base + (size_t)t * H + j] = hn;
    }
    __syncthreads();
  }
}

// ---------------------------------------------------------------------------
// Host-side orchestration (all on `stream`; graph-capture safe).
// ---------------------------------------------------------------------------
extern "C" void kernel_launch(void* const* d_in, const int* in_sizes, int n_in,
                              void* d_out, int out_size, void* d_ws, size_t ws_size,
                              hipStream_t stream) {
  const int B = 8, T = 1024, D = 128, H = 128, L = 2;
  const float* input = (const float*)d_in[0];
  const float* Wq = (const float*)d_in[1];  const float* bq = (const float*)d_in[2];
  const float* Wk = (const float*)d_in[3];  const float* bk = (const float*)d_in[4];
  const float* Wv = (const float*)d_in[5];  const float* bv = (const float*)d_in[6];
  const float* Wi = (const float*)d_in[7];  const float* bi = (const float*)d_in[8];
  const float* Wf = (const float*)d_in[9];  const float* bfp = (const float*)d_in[10];
  const float* Wo = (const float*)d_in[11]; const float* bo = (const float*)d_in[12];
  float* out = (float*)d_out;
  float* ws  = (float*)d_ws;

  const size_t NTH = (size_t)B * T * H;
  float* Qb    = ws + 0 * NTH;
  float* Kb    = ws + 1 * NTH;
  float* Vb    = ws + 2 * NTH;
  float* Igb   = ws + 3 * NTH;
  float* Fgb   = ws + 4 * NTH;
  float* attnb = ws + 5 * NTH;
  float* xb    = ws + 6 * NTH;

  const int Mrows = B * T;
  dim3 blk(256);

  proj_wmma<<<Mrows / 16, blk, 0, stream>>>(input, Wq, bq, Qb, 0);
  proj_wmma<<<Mrows / 16, blk, 0, stream>>>(input, Wk, bk, Kb, 0);
  proj_wmma<<<Mrows / 16, blk, 0, stream>>>(input, Wv, bv, Vb, 0);

  for (int l = 0; l < L; ++l) {
    const float* x = (l == 0) ? input : xb;
    proj_wmma<<<Mrows / 16, blk, 0, stream>>>(x, Wi + (size_t)l * D * H, bi + (size_t)l * H, Igb, 1);
    proj_wmma<<<Mrows / 16, blk, 0, stream>>>(x, Wf + (size_t)l * D * H, bfp + (size_t)l * H, Fgb, 1);
    mlstm_chunk_scan<<<dim3(B, H / 16), blk, 0, stream>>>(Qb, Kb, Vb, Igb, Fgb, attnb, T);
    gate_scan<<<B, blk, 0, stream>>>(attnb, Wo + (size_t)l * H * H, bo + (size_t)l * H,
                                     (l == 0) ? xb : out, T);
  }
  (void)in_sizes; (void)n_in; (void)out_size; (void)ws_size;
}
